// SimpleMHA_17059610100298
// MI455X (gfx1250) — matrix operand order repaired and checked by reading, not compiled, not run on hardware
//
#include <hip/hip_runtime.h>

// ---------------------------------------------------------------------------
// SimpleMHA forward on MI455X (gfx1250, wave32, WMMA f32_16x16x32_f16)
// Pipeline: QKV proj GEMMs -> flash attention -> output proj GEMM
// Round 2: software-pipelined GEMM (register prefetch of next K-tile) and
// async-to-LDS double-buffered K staging in attention (ASYNCcnt path).
// Workspace layout (needs 64 MB):
//   [0,16MB)   qh  f16 [B][H][S][64]
//   [16,32MB)  kh  f16
//   [32,48MB)  vh  f16
//   [48,64MB)  ctx f16 [B][S][1024]
// ---------------------------------------------------------------------------

typedef __attribute__((ext_vector_type(16))) _Float16 v16h;
typedef __attribute__((ext_vector_type(8)))  _Float16 v8h;
typedef __attribute__((ext_vector_type(8)))  float    v8f;

#define D_MODEL  1024
#define N_HEADS  16
#define HEAD_DIM 64
#define BATCH    4
#define SEQ      2048
#define M_TOTAL  (BATCH * SEQ)   // 8192

__device__ __forceinline__ v16h mk16(v8h lo, v8h hi) {
  union { v16h v; v8h h[2]; } u;
  u.h[0] = lo; u.h[1] = hi;
  return u.v;
}

__device__ __forceinline__ v8f wmma_f16(v16h a, v16h b, v8f c) {
  // D = A(16x32 f16) x B(32x16 f16) + C(16x16 f32)
  return __builtin_amdgcn_wmma_f32_16x16x32_f16(false, a, false, b,
                                                (short)0, c, false, false);
}

// Async copy 16B global -> LDS (GLOBAL_LOAD_ASYNC_TO_LDS_B128, ASYNCcnt).
// LDS destination = low 32 bits of the generic shared pointer (aperture rule:
// the LDS offset is taken from the address's low 32 bits).
__device__ __forceinline__ void async_ld_b128(const void* g, void* lds) {
  unsigned loff = (unsigned)(unsigned long long)lds;
  asm volatile("global_load_async_to_lds_b128 %0, %1, off"
               :: "v"(loff), "v"(g) : "memory");
}
__device__ __forceinline__ void wait_async0() {
  asm volatile("s_wait_asynccnt 0x0" ::: "memory");
}

__device__ __forceinline__ v8h pack8(float4 f0, float4 f1) {
  v8h hh;
  hh[0]=(_Float16)f0.x; hh[1]=(_Float16)f0.y;
  hh[2]=(_Float16)f0.z; hh[3]=(_Float16)f0.w;
  hh[4]=(_Float16)f1.x; hh[5]=(_Float16)f1.y;
  hh[6]=(_Float16)f1.z; hh[7]=(_Float16)f1.w;
  return hh;
}

// ===========================================================================
// GEMM: out[M,N] = A[M,K] @ W[N,K]^T + bias[N]
//   mode 0: A = f32 global, output f16 in head layout [B][H][S][64]
//   mode 1: A = f16 (ctx workspace), output f32 flat [M][N]
// 128x128 tile, K-step 64, 256 threads = 8 waves (4 m x 2 n), 2x4 WMMA accs.
// Software pipelined: next K-tile is fetched into registers while the WMMAs
// consume the current tile from LDS.
// ===========================================================================
#define GBM 128
#define GBN 128
#define GBK 64
#define GSTR (GBK + 8)   // LDS row stride (halfs), 144 B -> 16B aligned

__global__ __launch_bounds__(256)
void gemm_wmma_kernel(const float* __restrict__ A32,
                      const _Float16* __restrict__ A16,
                      const float* __restrict__ W,
                      const float* __restrict__ bias,
                      float* __restrict__ outF,
                      _Float16* __restrict__ outH,
                      int mode) {
  __shared__ _Float16 Asl[GBM * GSTR];
  __shared__ _Float16 Bsl[GBN * GSTR];

  const int tid  = threadIdx.x;
  const int wid  = tid >> 5;
  const int lane = tid & 31;
  const int lh   = lane >> 4;    // half-wave index (0/1)
  const int ln   = lane & 15;    // lane within half
  const int wm   = wid >> 1;     // 0..3
  const int wn   = wid & 1;      // 0..1
  const int m_block = blockIdx.y * GBM;
  const int n_block = blockIdx.x * GBN;

  const v8f zf = {0.f,0.f,0.f,0.f,0.f,0.f,0.f,0.f};
  v8f acc[2][4];
#pragma unroll
  for (int im = 0; im < 2; ++im)
#pragma unroll
    for (int in_ = 0; in_ < 4; ++in_) acc[im][in_] = zf;

  const int lrow = tid >> 1;         // 0..127 (tile row this thread stages)
  const int lcb  = (tid & 1) * 32;   // k-column base 0/32

  float4 ar[8];   // prefetched A slice (f32 path)
  v8h    ah[4];   // prefetched A slice (f16 path)
  float4 wr[8];   // prefetched W slice

  auto load_regs = [&](int k0) {
    if (mode == 0) {
      const float4* gp =
          (const float4*)(A32 + (size_t)(m_block + lrow) * D_MODEL + k0 + lcb);
#pragma unroll
      for (int i = 0; i < 8; ++i) ar[i] = gp[i];
    } else {
      const v8h* gp =
          (const v8h*)(A16 + (size_t)(m_block + lrow) * D_MODEL + k0 + lcb);
#pragma unroll
      for (int i = 0; i < 4; ++i) ah[i] = gp[i];
    }
    const float4* wp =
        (const float4*)(W + (size_t)(n_block + lrow) * D_MODEL + k0 + lcb);
#pragma unroll
    for (int i = 0; i < 8; ++i) wr[i] = wp[i];
  };

  load_regs(0);

  for (int k0 = 0; k0 < D_MODEL; k0 += GBK) {
    // ---- stage prefetched registers into LDS (convert f32->f16) ----
    if (mode == 0) {
#pragma unroll
      for (int i = 0; i < 4; ++i)
        *(v8h*)&Asl[lrow * GSTR + lcb + i * 8] = pack8(ar[2*i], ar[2*i+1]);
    } else {
#pragma unroll
      for (int i = 0; i < 4; ++i)
        *(v8h*)&Asl[lrow * GSTR + lcb + i * 8] = ah[i];
    }
#pragma unroll
    for (int i = 0; i < 4; ++i)
      *(v8h*)&Bsl[lrow * GSTR + lcb + i * 8] = pack8(wr[2*i], wr[2*i+1]);
    __syncthreads();

    // ---- issue next tile's global loads; they retire during the WMMAs ----
    if (k0 + GBK < D_MODEL) load_regs(k0 + GBK);

    // ---- WMMA over the two 16x16x32 K-steps in this stage ----
#pragma unroll
    for (int ks = 0; ks < 2; ++ks) {
      const int kb = ks * 32;
      v16h af[2], bf[4];
#pragma unroll
      for (int im = 0; im < 2; ++im) {
        // A layout: lane=M; half 0 -> K {kb..+7, kb+16..+23}, half 1 -> {+8,+24}
        const _Float16* arp = &Asl[(wm * 32 + im * 16 + ln) * GSTR + kb];
        af[im] = mk16(*(const v8h*)(arp + lh * 8),
                      *(const v8h*)(arp + 16 + lh * 8));
      }
#pragma unroll
      for (int in_ = 0; in_ < 4; ++in_) {
        // B layout: lane=N; half 0 -> K {kb..+7, kb+16..+23}, half 1 -> {+8,+24}
        const _Float16* brp =
            &Bsl[(wn * 64 + in_ * 16 + ln) * GSTR + kb + lh * 8];
        bf[in_] = mk16(*(const v8h*)(brp), *(const v8h*)(brp + 16));
      }
#pragma unroll
      for (int im = 0; im < 2; ++im)
#pragma unroll
        for (int in_ = 0; in_ < 4; ++in_)
          acc[im][in_] = wmma_f16(af[im], bf[in_], acc[im][in_]);
    }
    __syncthreads();
  }

  // ---- epilogue: C layout lane=N, VGPR r -> M=r (+8 for upper half) ----
#pragma unroll
  for (int im = 0; im < 2; ++im) {
#pragma unroll
    for (int in_ = 0; in_ < 4; ++in_) {
      const int n = n_block + wn * 64 + in_ * 16 + ln;
      const float bv = bias[n];
#pragma unroll
      for (int r = 0; r < 8; ++r) {
        const int m = m_block + wm * 32 + im * 16 + r + lh * 8;
        const float val = acc[im][in_][r] + bv;
        if (mode == 0) {
          const int bb = m >> 11, q = m & (SEQ - 1);
          const int hh_ = n >> 6, d = n & 63;
          outH[(((size_t)bb * N_HEADS + hh_) * SEQ + q) * HEAD_DIM + d] =
              (_Float16)val;
        } else {
          outF[(size_t)m * D_MODEL + n] = val;
        }
      }
    }
  }
}

// ===========================================================================
// Flash attention: one (b,h) per blockIdx.y, 128 q rows per block.
// 8 waves x 16 q-rows; kv tiles of 64 with online softmax.
// K tile: double-buffered LDS filled by GLOBAL_LOAD_ASYNC_TO_LDS_B128.
// V tile: register-prefetched, transposed into LDS [d][kv].
// ===========================================================================
#define KV_TILE 64
#define KSTR2   (HEAD_DIM + 8)   // 72 halfs, 144 B stride
#define N_KV_IT (SEQ / KV_TILE)  // 32

__global__ __launch_bounds__(256)
void attn_wmma_kernel(const _Float16* __restrict__ qh,
                      const _Float16* __restrict__ kh,
                      const _Float16* __restrict__ vh,
                      const unsigned char* __restrict__ kv_mask,
                      _Float16* __restrict__ ctx) {
  __shared__ _Float16 Kl[2][KV_TILE * KSTR2];   // [buf][kv][d]
  __shared__ _Float16 Vt[HEAD_DIM * KSTR2];     // [d][kv]  (transposed V)
  __shared__ _Float16 Pl[8 * 16 * KSTR2];       // per-wave P staging [16][72]

  const int tid  = threadIdx.x;
  const int wid  = tid >> 5;
  const int lane = tid & 31;
  const int lh   = lane >> 4;
  const int ln   = lane & 15;

  const int bh = blockIdx.y;
  const int b  = bh >> 4;
  const int h  = bh & (N_HEADS - 1);
  const _Float16* Qp = qh + (size_t)bh * SEQ * HEAD_DIM;
  const _Float16* Kp = kh + (size_t)bh * SEQ * HEAD_DIM;
  const _Float16* Vp = vh + (size_t)bh * SEQ * HEAD_DIM;
  const unsigned char* mrow = kv_mask + (size_t)b * SEQ;

  const int q0 = blockIdx.x * 128 + wid * 16;

  // cooperative-load coordinates: each thread owns 32B of one kv row
  const int cr  = tid >> 2;          // 0..63 kv row
  const int cc0 = (tid & 3) * 16;    // d chunk base (halfs)

  // ---- Q fragments (A layout), kept in registers all kernel ----
  v16h qf[2];
  {
    const _Float16* qr = Qp + (size_t)(q0 + ln) * HEAD_DIM;
#pragma unroll
    for (int ks = 0; ks < 2; ++ks) {
      const int kb = ks * 32;
      qf[ks] = mk16(*(const v8h*)(qr + kb + lh * 8),
                    *(const v8h*)(qr + kb + 16 + lh * 8));
    }
  }

  const v8f zf = {0.f,0.f,0.f,0.f,0.f,0.f,0.f,0.f};
  v8f o[4];
#pragma unroll
  for (int dt = 0; dt < 4; ++dt) o[dt] = zf;
  float m_r[8], l_r[8];
#pragma unroll
  for (int r = 0; r < 8; ++r) { m_r[r] = -1e30f; l_r[r] = 0.f; }

  _Float16* Pb = &Pl[wid * 16 * KSTR2];

  // ---- prologue: async K tile 0, register-load V tile 0 ----
  v8h vreg0, vreg1;
  {
    const _Float16* kg = Kp + (size_t)cr * HEAD_DIM + cc0;
    async_ld_b128(kg,     &Kl[0][cr * KSTR2 + cc0]);
    async_ld_b128(kg + 8, &Kl[0][cr * KSTR2 + cc0 + 8]);
    const v8h* vg = (const v8h*)(Vp + (size_t)cr * HEAD_DIM + cc0);
    vreg0 = vg[0]; vreg1 = vg[1];
  }

  for (int it = 0; it < N_KV_IT; ++it) {
    const int kv0 = it * KV_TILE;
    const int cur = it & 1, nxt = cur ^ 1;

    wait_async0();        // our async K chunks (current tile) have landed
    __syncthreads();      // everyone's landed; prior reads of Vt/Kl[nxt] done

    // ---- stage V (transpose) from registers into LDS ----
#pragma unroll
    for (int i = 0; i < 8; ++i) Vt[(cc0 + i) * KSTR2 + cr]     = vreg0[i];
#pragma unroll
    for (int i = 0; i < 8; ++i) Vt[(cc0 + 8 + i) * KSTR2 + cr] = vreg1[i];

    // ---- prefetch next tile (async K -> Kl[nxt], V -> regs) ----
    if (it + 1 < N_KV_IT) {
      const int kvn = kv0 + KV_TILE;
      const _Float16* kg = Kp + (size_t)(kvn + cr) * HEAD_DIM + cc0;
      async_ld_b128(kg,     &Kl[nxt][cr * KSTR2 + cc0]);
      async_ld_b128(kg + 8, &Kl[nxt][cr * KSTR2 + cc0 + 8]);
      const v8h* vg = (const v8h*)(Vp + (size_t)(kvn + cr) * HEAD_DIM + cc0);
      vreg0 = vg[0]; vreg1 = vg[1];
    }
    __syncthreads();      // Vt visible to all waves

    // ---- S = Q K^T for this wave's 16 rows x 64 kv ----
    const _Float16* Kb = &Kl[cur][0];
    v8f s[4];
#pragma unroll
    for (int t = 0; t < 4; ++t) {
      v8f a = zf;
#pragma unroll
      for (int ks = 0; ks < 2; ++ks) {
        const _Float16* kr = &Kb[(t * 16 + ln) * KSTR2 + ks * 32 + lh * 8];
        a = wmma_f16(qf[ks], mk16(*(const v8h*)kr, *(const v8h*)(kr + 16)), a);
      }
      s[t] = a;
    }

    // ---- scale + mask (reference: masked -> exactly -10000) ----
    bool mk[4];
#pragma unroll
    for (int t = 0; t < 4; ++t) mk[t] = mrow[kv0 + t * 16 + ln] != 0;
#pragma unroll
    for (int t = 0; t < 4; ++t)
#pragma unroll
      for (int r = 0; r < 8; ++r)
        s[t][r] = mk[t] ? s[t][r] * 0.125f : -10000.0f;

    // ---- online softmax: new row max (reduce over 16-lane half) ----
    float mnew[8], alpha[8];
#pragma unroll
    for (int r = 0; r < 8; ++r) {
      float v = fmaxf(fmaxf(s[0][r], s[1][r]), fmaxf(s[2][r], s[3][r]));
      v = fmaxf(v, __shfl_xor(v, 1, 32));
      v = fmaxf(v, __shfl_xor(v, 2, 32));
      v = fmaxf(v, __shfl_xor(v, 4, 32));
      v = fmaxf(v, __shfl_xor(v, 8, 32));
      mnew[r]  = fmaxf(v, m_r[r]);
      alpha[r] = __expf(m_r[r] - mnew[r]);
      m_r[r]   = mnew[r];
    }

    // ---- P = exp(S - m), row sums, stage P to LDS in A layout [M][kv] ----
    float rs[8];
#pragma unroll
    for (int r = 0; r < 8; ++r) rs[r] = 0.f;
#pragma unroll
    for (int t = 0; t < 4; ++t)
#pragma unroll
      for (int r = 0; r < 8; ++r) {
        const float p = __expf(s[t][r] - mnew[r]);
        rs[r] += p;
        Pb[(r + lh * 8) * KSTR2 + t * 16 + ln] = (_Float16)p;
      }
#pragma unroll
    for (int r = 0; r < 8; ++r) {
      float v = rs[r];
      v += __shfl_xor(v, 1, 32);
      v += __shfl_xor(v, 2, 32);
      v += __shfl_xor(v, 4, 32);
      v += __shfl_xor(v, 8, 32);
      l_r[r] = l_r[r] * alpha[r] + v;
    }

    // P staging is wave-private; drain our DS stores before re-reading.
    asm volatile("s_wait_dscnt 0" ::: "memory");

    // ---- O = O*alpha + P @ V ----
#pragma unroll
    for (int dt = 0; dt < 4; ++dt)
#pragma unroll
      for (int r = 0; r < 8; ++r) o[dt][r] *= alpha[r];
#pragma unroll
    for (int ks = 0; ks < 2; ++ks) {
      const _Float16* pr = Pb + ln * KSTR2 + ks * 32;   // A layout, lane=M
      v16h pf = mk16(*(const v8h*)(pr + lh * 8),
                     *(const v8h*)(pr + 16 + lh * 8));
#pragma unroll
      for (int dt = 0; dt < 4; ++dt) {
        const _Float16* vr = &Vt[(dt * 16 + ln) * KSTR2 + ks * 32 + lh * 8];
        o[dt] = wmma_f16(pf, mk16(*(const v8h*)vr, *(const v8h*)(vr + 16)),
                         o[dt]);
      }
    }
  }

  // ---- epilogue: ctx[b][q][h*64+d] = O / l ----
#pragma unroll
  for (int dt = 0; dt < 4; ++dt) {
#pragma unroll
    for (int r = 0; r < 8; ++r) {
      const int q = q0 + r + lh * 8;
      const int d = dt * 16 + ln;
      const float val = o[dt][r] / l_r[r];
      ctx[((size_t)b * SEQ + q) * D_MODEL + h * HEAD_DIM + d] = (_Float16)val;
    }
  }
}

// ===========================================================================
extern "C" void kernel_launch(void* const* d_in, const int* in_sizes, int n_in,
                              void* d_out, int out_size, void* d_ws,
                              size_t ws_size, hipStream_t stream) {
  const float* q  = (const float*)d_in[0];
  const float* k  = (const float*)d_in[1];
  const float* v  = (const float*)d_in[2];
  const unsigned char* kvm = (const unsigned char*)d_in[3];
  const float* Wq = (const float*)d_in[4];
  const float* bq = (const float*)d_in[5];
  const float* Wk = (const float*)d_in[6];
  const float* bk = (const float*)d_in[7];
  const float* Wv = (const float*)d_in[8];
  const float* bv = (const float*)d_in[9];
  const float* Wo = (const float*)d_in[10];
  const float* bo = (const float*)d_in[11];
  float* out = (float*)d_out;

  char* ws = (char*)d_ws;
  const size_t SEG = (size_t)16 * 1024 * 1024;   // 16 MB per f16 tensor
  _Float16* qh  = (_Float16*)(ws);
  _Float16* kh  = (_Float16*)(ws + SEG);
  _Float16* vh  = (_Float16*)(ws + 2 * SEG);
  _Float16* ctx = (_Float16*)(ws + 3 * SEG);

  dim3 gg(D_MODEL / GBN, M_TOTAL / GBM);   // (8, 64)
  gemm_wmma_kernel<<<gg, 256, 0, stream>>>(q, nullptr, Wq, bq, nullptr, qh, 0);
  gemm_wmma_kernel<<<gg, 256, 0, stream>>>(k, nullptr, Wk, bk, nullptr, kh, 0);
  gemm_wmma_kernel<<<gg, 256, 0, stream>>>(v, nullptr, Wv, bv, nullptr, vh, 0);

  attn_wmma_kernel<<<dim3(SEQ / 128, BATCH * N_HEADS), 256, 0, stream>>>(
      qh, kh, vh, kvm, ctx);

  gemm_wmma_kernel<<<gg, 256, 0, stream>>>(nullptr, ctx, Wo, bo, out, nullptr, 1);
}
